// UPGAT_63196148793599
// MI455X (gfx1250) — matrix operand order repaired and checked by reading, not compiled, not run on hardware
//
#include <hip/hip_runtime.h>
#include <hip/hip_bf16.h>

typedef __bf16 bf16_t;
typedef __attribute__((ext_vector_type(16))) __bf16 v16bf;
typedef __attribute__((ext_vector_type(8)))  __bf16 v8bf;
typedef __attribute__((ext_vector_type(8)))  float   v8f;

#define DIM 512
#define EPSV 1e-12f

__device__ __forceinline__ bf16_t f2bf(float f) {
    unsigned u = __builtin_bit_cast(unsigned, f);
    u += 0x7FFFu + ((u >> 16) & 1u);                 // round-to-nearest-even
    unsigned short h = (unsigned short)(u >> 16);
    return __builtin_bit_cast(bf16_t, h);
}
__device__ __forceinline__ float bf2f(bf16_t b) {
    unsigned short s = __builtin_bit_cast(unsigned short, b);
    unsigned u = ((unsigned)s) << 16;
    return __builtin_bit_cast(float, u);
}
__device__ __forceinline__ v16bf mk16(v8bf lo, v8bf hi) {
    v16bf r;
#pragma unroll
    for (int i = 0; i < 8; i++) { r[i] = lo[i]; r[i + 8] = hi[i]; }
    return r;
}
__device__ __forceinline__ v8f vzero() {
    v8f z;
#pragma unroll
    for (int i = 0; i < 8; i++) z[i] = 0.f;
    return z;
}
// A-fragment (16x32 bf16): lane<16 holds K=kb+0..7 & kb+16..23; lane>=16 K=kb+8..15 & kb+24..31
__device__ __forceinline__ v16bf load_a(const bf16_t* rowp, int kb, int half) {
    v8bf lo = *(const v8bf*)(rowp + kb + 8 * half);
    v8bf hi = *(const v8bf*)(rowp + kb + 16 + 8 * half);
    return mk16(lo, hi);
}
// B-fragment (32x16 bf16): col = lane%16, lane<16 holds K=kb+0..7 & kb+16..23; lane>=16 K=kb+8..15 & kb+24..31
__device__ __forceinline__ v16bf load_b(const bf16_t* rowp, int kb, int half) {
    v8bf lo = *(const v8bf*)(rowp + kb + 8 * half);
    v8bf hi = *(const v8bf*)(rowp + kb + 16 + 8 * half);
    return mk16(lo, hi);
}
// reduce across the 16 lanes of a half-wave (masks < 16 stay within the half)
__device__ __forceinline__ float red16(float x) {
    x += __shfl_xor(x, 1, 32);
    x += __shfl_xor(x, 2, 32);
    x += __shfl_xor(x, 4, 32);
    x += __shfl_xor(x, 8, 32);
    return x;
}

// ---------------- e0 = l2norm(ent_emb_init), stored bf16 ----------------
__global__ __launch_bounds__(256) void k_norm_e0(const float* __restrict__ ent,
                                                 bf16_t* __restrict__ e0, int N) {
    __shared__ float red[256];
    int n = blockIdx.x;
    const float* row = ent + (size_t)n * DIM;
    float s = 0.f;
    for (int i = threadIdx.x; i < DIM; i += 256) { float v = row[i]; s += v * v; }
    red[threadIdx.x] = s;
    __syncthreads();
    for (int off = 128; off > 0; off >>= 1) {
        if (threadIdx.x < off) red[threadIdx.x] += red[threadIdx.x + off];
        __syncthreads();
    }
    float scale = 1.f / fmaxf(sqrtf(red[0]), EPSV);
    for (int i = threadIdx.x; i < DIM; i += 256)
        e0[(size_t)n * DIM + i] = f2bf(row[i] * scale);
}

// ---------------- weight prep: W1 row-major bf16; W_E / W_R transposed bf16 ----------------
__global__ __launch_bounds__(256) void k_prep_w(const float* __restrict__ W1,
                                                const float* __restrict__ WE,
                                                const float* __restrict__ WR,
                                                bf16_t* __restrict__ W1bf,
                                                bf16_t* __restrict__ WEbt,
                                                bf16_t* __restrict__ WRbt) {
    int i = blockIdx.x * 256 + threadIdx.x;     // 0 .. 512*512-1
    int d = i >> 9, k = i & 511;
    W1bf[i] = f2bf(W1[i]);                      // B[k][n] = W1[n][k] -> row n contiguous in k
    WEbt[i] = f2bf(WE[k * DIM + d]);            // B[k][d] = WE[k][d] -> store row d over k
    WRbt[i] = f2bf(WR[k * DIM + d]);
}

// ---------------- fused edge kernel: 16 edges/block, dual WMMA GEMM + attention + scatter ----
__global__ __launch_bounds__(128) void k_edge(
    const int* __restrict__ nl, const int* __restrict__ el,
    const bf16_t* __restrict__ e0, const float* __restrict__ rel_emb,
    const bf16_t* __restrict__ W1bf, const float* __restrict__ Wa,
    const float* __restrict__ g0,
    float* __restrict__ denom_sum, float* __restrict__ denom_set,
    float* __restrict__ agg_sum, float* __restrict__ agg_set, int E) {
    __shared__ bf16_t sX[16][DIM];      // h_j * g_k
    __shared__ bf16_t sXb[16][DIM];     // h_i * g_0
    __shared__ bf16_t sHW[16][DIM];     // h_i * W_a
    __shared__ int    sSrc[16];
    __shared__ int    sVal[16];
    __shared__ float  sRed[4][16], sRedB[4][16];
    __shared__ float  sExp[16], sExpB[16];

    const int tid = threadIdx.x;
    const int tile = blockIdx.x;

    for (int r = 0; r < 16; r++) {
        int e = tile * 16 + r;
        int ok = (e < E) ? 1 : 0;
        int ec = ok ? e : 0;
        int src = nl[ec], dst = nl[(size_t)E + ec], rl = el[ec];
        if (tid == 0) { sSrc[r] = src; sVal[r] = ok; }
        const bf16_t* hi = e0 + (size_t)src * DIM;
        const bf16_t* hj = e0 + (size_t)dst * DIM;
        const float* gk = rel_emb + (size_t)rl * DIM;
#pragma unroll
        for (int j = 0; j < 4; j++) {
            int c = tid + 128 * j;
            float hif = ok ? bf2f(hi[c]) : 0.f;
            float hjf = ok ? bf2f(hj[c]) : 0.f;
            sX[r][c]  = f2bf(hjf * gk[c]);
            sXb[r][c] = f2bf(hif * g0[c]);
            sHW[r][c] = f2bf(hif * Wa[c]);
        }
    }
    __syncthreads();

    const int wave = tid >> 5, lane = tid & 31;
    const int half = lane >> 4, m = lane & 15;
    const int colbase = wave * 128;

    v8f acc[8], accB[8];
#pragma unroll
    for (int t = 0; t < 8; t++) { acc[t] = vzero(); accB[t] = vzero(); }

    for (int kb = 0; kb < DIM; kb += 32) {
        v16bf aX  = load_a(&sX[m][0], kb, half);
        v16bf aXb = load_a(&sXb[m][0], kb, half);
#pragma unroll
        for (int t = 0; t < 8; t++) {
            const bf16_t* bp = W1bf + (size_t)(colbase + t * 16 + m) * DIM;
            v16bf bF = load_b(bp, kb, half);
            acc[t]  = __builtin_amdgcn_wmma_f32_16x16x32_bf16(false, aX,  false, bF,
                                                              (short)0, acc[t],  false, false);
            accB[t] = __builtin_amdgcn_wmma_f32_16x16x32_bf16(false, aXb, false, bF,
                                                              (short)0, accB[t], false, false);
        }
    }

    // attention logits: sum_d (h_i*W_a)[d] * tanh(c[d]) per edge row
    float rp[8], rpB[8];
#pragma unroll
    for (int v = 0; v < 8; v++) { rp[v] = 0.f; rpB[v] = 0.f; }
#pragma unroll
    for (int t = 0; t < 8; t++) {
        int col = colbase + t * 16 + m;
#pragma unroll
        for (int v = 0; v < 8; v++) {
            float hw = bf2f(sHW[v + 8 * half][col]);
            rp[v]  += hw * tanhf(acc[t][v]);
            rpB[v] += hw * tanhf(accB[t][v]);
        }
    }
#pragma unroll
    for (int v = 0; v < 8; v++) { rp[v] = red16(rp[v]); rpB[v] = red16(rpB[v]); }
    if (m == 0) {
#pragma unroll
        for (int v = 0; v < 8; v++) {
            sRed[wave][v + 8 * half]  = rp[v];
            sRedB[wave][v + 8 * half] = rpB[v];
        }
    }
    __syncthreads();

    if (tid < 16) {
        float x  = sRed[0][tid] + sRed[1][tid] + sRed[2][tid] + sRed[3][tid];
        float xb = sRedB[0][tid] + sRedB[1][tid] + sRedB[2][tid] + sRedB[3][tid];
        float l  = x > 0.f ? x : 0.2f * x;        // leaky_relu(x, 0.2)
        float lb = xb > 0.f ? xb : 0.2f * xb;
        float aexp  = __expf(-l);
        float abexp = __expf(-lb);
        if (!sVal[tid]) { aexp = 0.f; abexp = 0.f; }
        sExp[tid] = aexp; sExpB[tid] = abexp;
        if (sVal[tid]) {
            atomicAdd(&denom_sum[sSrc[tid]], aexp);
            denom_set[sSrc[tid]] = abexp;          // duplicate src write identical value
        }
    }
    __syncthreads();

    // scatter: agg_sum[src] += a_exp*c ; agg_set[src] = ab_exp*c_base
#pragma unroll
    for (int t = 0; t < 8; t++) {
        int col = colbase + t * 16 + m;
#pragma unroll
        for (int v = 0; v < 8; v++) {
            int r = v + 8 * half;
            if (sVal[r]) {
                size_t o = (size_t)sSrc[r] * DIM + col;
                atomicAdd(&agg_sum[o], sExp[r] * acc[t][v]);
                agg_set[o] = sExpB[r] * accB[t][v];
            }
        }
    }
}

// ---------------- h_prime = elu((agg_set+agg_sum)/denom), in-place into agg_sum ----------------
__global__ __launch_bounds__(256) void k_hp(const float* __restrict__ denom_sum,
                                            const float* __restrict__ denom_set,
                                            const float* __restrict__ agg_set,
                                            float* __restrict__ agg_sum_io,
                                            unsigned long long total) {
    unsigned long long i = (unsigned long long)blockIdx.x * 256u + threadIdx.x;
    unsigned long long stride = (unsigned long long)gridDim.x * 256u;
    for (; i < total; i += stride) {
        unsigned long long n = i >> 9;
        float den = denom_set[n] + denom_sum[n];
        den = (den == 0.f) ? EPSV : den;
        float v = (agg_set[i] + agg_sum_io[i]) / den;
        agg_sum_io[i] = (v > 0.f) ? v : (__expf(v) - 1.f);   // elu
    }
}

// ---------------- ent_new = l2norm(e0 @ W_E + h_prime) : WMMA + fused residual + row norm ------
__global__ __launch_bounds__(128) void k_ent(const bf16_t* __restrict__ e0,
                                             const bf16_t* __restrict__ WEbt,
                                             const float* __restrict__ hprime,
                                             float* __restrict__ ent_new, int N) {
    __shared__ bf16_t sE[16][DIM];
    __shared__ float sRed[4][16];
    __shared__ float sS[16];
    const int tid = threadIdx.x, tile = blockIdx.x;
    const size_t row0 = (size_t)tile * 16;

    for (int i = tid; i < 16 * DIM; i += 128) {
        size_t r = row0 + (i >> 9);
        ((bf16_t*)sE)[i] = (r < (size_t)N) ? e0[r * DIM + (i & 511)] : f2bf(0.f);
    }
    __syncthreads();

    const int wave = tid >> 5, lane = tid & 31;
    const int half = lane >> 4, m = lane & 15;
    const int colbase = wave * 128;

    v8f acc[8];
#pragma unroll
    for (int t = 0; t < 8; t++) acc[t] = vzero();

    for (int kb = 0; kb < DIM; kb += 32) {
        v16bf aF = load_a(&sE[m][0], kb, half);
#pragma unroll
        for (int t = 0; t < 8; t++) {
            const bf16_t* bp = WEbt + (size_t)(colbase + t * 16 + m) * DIM;
            v16bf bF = load_b(bp, kb, half);
            acc[t] = __builtin_amdgcn_wmma_f32_16x16x32_bf16(false, aF, false, bF,
                                                             (short)0, acc[t], false, false);
        }
    }

    float rp[8];
#pragma unroll
    for (int v = 0; v < 8; v++) rp[v] = 0.f;
#pragma unroll
    for (int t = 0; t < 8; t++) {
        int col = colbase + t * 16 + m;
#pragma unroll
        for (int v = 0; v < 8; v++) {
            size_t node = row0 + v + 8 * half;
            float val = acc[t][v] + ((node < (size_t)N) ? hprime[node * DIM + col] : 0.f);
            acc[t][v] = val;
            rp[v] += val * val;
        }
    }
#pragma unroll
    for (int v = 0; v < 8; v++) rp[v] = red16(rp[v]);
    if (m == 0) {
#pragma unroll
        for (int v = 0; v < 8; v++) sRed[wave][v + 8 * half] = rp[v];
    }
    __syncthreads();
    if (tid < 16) sS[tid] = sRed[0][tid] + sRed[1][tid] + sRed[2][tid] + sRed[3][tid];
    __syncthreads();

#pragma unroll
    for (int t = 0; t < 8; t++) {
        int col = colbase + t * 16 + m;
#pragma unroll
        for (int v = 0; v < 8; v++) {
            int r = v + 8 * half;
            size_t node = row0 + r;
            if (node < (size_t)N) {
                float sc = 1.f / fmaxf(sqrtf(sS[r]), EPSV);
                ent_new[node * DIM + col] = acc[t][v] * sc;
            }
        }
    }
}

// ---------------- rel_new = rel_emb @ W_R ----------------
__global__ __launch_bounds__(128) void k_rel(const float* __restrict__ rel_emb,
                                             const bf16_t* __restrict__ WRbt,
                                             float* __restrict__ rel_new, int R) {
    __shared__ bf16_t sA[16][DIM];
    const int tid = threadIdx.x, tile = blockIdx.x;
    const int row0 = tile * 16;
    for (int i = tid; i < 16 * DIM; i += 128) {
        int r = row0 + (i >> 9), c = i & 511;
        ((bf16_t*)sA)[i] = f2bf((r < R) ? rel_emb[(size_t)r * DIM + c] : 0.f);
    }
    __syncthreads();

    const int wave = tid >> 5, lane = tid & 31;
    const int half = lane >> 4, m = lane & 15;
    const int colbase = wave * 128;
    v8f acc[8];
#pragma unroll
    for (int t = 0; t < 8; t++) acc[t] = vzero();

    for (int kb = 0; kb < DIM; kb += 32) {
        v16bf aF = load_a(&sA[m][0], kb, half);
#pragma unroll
        for (int t = 0; t < 8; t++) {
            const bf16_t* bp = WRbt + (size_t)(colbase + t * 16 + m) * DIM;
            v16bf bF = load_b(bp, kb, half);
            acc[t] = __builtin_amdgcn_wmma_f32_16x16x32_bf16(false, aF, false, bF,
                                                             (short)0, acc[t], false, false);
        }
    }
#pragma unroll
    for (int t = 0; t < 8; t++) {
        int col = colbase + t * 16 + m;
#pragma unroll
        for (int v = 0; v < 8; v++) {
            int r = row0 + v + 8 * half;
            if (r < R) rel_new[(size_t)r * DIM + col] = acc[t][v];
        }
    }
}

// ---------------- output gather [B,3,D] ----------------
__global__ __launch_bounds__(256) void k_out(const int* __restrict__ triples,
                                             const float* __restrict__ ent_new,
                                             const float* __restrict__ rel_new,
                                             float* __restrict__ out, int B) {
    int b = blockIdx.x;
    for (int i = threadIdx.x; i < 3 * DIM; i += 256) {
        int part = i >> 9, c = i & 511;
        int idx = triples[b * 3 + part];
        const float* srcp = (part == 1) ? (rel_new + (size_t)idx * DIM)
                                        : (ent_new + (size_t)idx * DIM);
        out[(size_t)b * (3 * DIM) + i] = srcp[c];
    }
}

extern "C" void kernel_launch(void* const* d_in, const int* in_sizes, int n_in,
                              void* d_out, int out_size, void* d_ws, size_t ws_size,
                              hipStream_t stream) {
    const int*   triples = (const int*)d_in[0];
    const int*   nl      = (const int*)d_in[1];   // [2,E]
    const int*   el      = (const int*)d_in[2];   // [E]
    const float* ent     = (const float*)d_in[3]; // [N,512]
    const float* rel     = (const float*)d_in[4]; // [R,512]
    const float* W1      = (const float*)d_in[5];
    const float* Wa      = (const float*)d_in[6];
    const float* g0      = (const float*)d_in[7];
    const float* WE      = (const float*)d_in[8];
    const float* WR      = (const float*)d_in[9];

    const int Bt = in_sizes[0] / 3;
    const int E  = in_sizes[2];
    const int N  = in_sizes[3] / DIM;
    const int R  = in_sizes[4] / DIM;

    char* ws = (char*)d_ws;
    size_t off = 0;
    auto walloc = [&](size_t bytes) {
        void* p = ws + off;
        off = (off + bytes + 255) & ~(size_t)255;
        return p;
    };
    bf16_t* e0bf  = (bf16_t*)walloc((size_t)N * DIM * 2);
    bf16_t* W1bf  = (bf16_t*)walloc((size_t)DIM * DIM * 2);
    bf16_t* WEbt  = (bf16_t*)walloc((size_t)DIM * DIM * 2);
    bf16_t* WRbt  = (bf16_t*)walloc((size_t)DIM * DIM * 2);
    float*  dsum  = (float*)walloc((size_t)N * 4);
    float*  dset  = (float*)walloc((size_t)N * 4);
    float*  agg_sum = (float*)walloc((size_t)N * DIM * 4);
    float*  agg_set = (float*)walloc((size_t)N * DIM * 4);
    float*  ent_new = (float*)walloc((size_t)N * DIM * 4);
    float*  rel_new = (float*)walloc((size_t)R * DIM * 4);

    hipMemsetAsync(dsum, 0, (size_t)N * 4, stream);
    hipMemsetAsync(dset, 0, (size_t)N * 4, stream);
    hipMemsetAsync(agg_sum, 0, (size_t)N * DIM * 4, stream);
    hipMemsetAsync(agg_set, 0, (size_t)N * DIM * 4, stream);

    k_prep_w<<<(DIM * DIM) / 256, 256, 0, stream>>>(W1, WE, WR, W1bf, WEbt, WRbt);
    k_norm_e0<<<N, 256, 0, stream>>>(ent, e0bf, N);
    k_edge<<<(E + 15) / 16, 128, 0, stream>>>(nl, el, e0bf, rel, W1bf, Wa, g0,
                                              dsum, dset, agg_sum, agg_set, E);
    k_hp<<<2048, 256, 0, stream>>>(dsum, dset, agg_set, agg_sum,
                                   (unsigned long long)N * DIM);
    k_ent<<<(N + 15) / 16, 128, 0, stream>>>(e0bf, WEbt, agg_sum, ent_new, N);
    k_rel<<<(R + 15) / 16, 128, 0, stream>>>(rel, WRbt, rel_new, R);
    k_out<<<Bt, 256, 0, stream>>>(triples, ent_new, rel_new, (float*)d_out, Bt);
}
